// SwinTiny_InternalRepresentation_20444044329734
// MI455X (gfx1250) — compile-verified, matrix operand order checked by reading, not run on hardware
//
#include <hip/hip_runtime.h>
#include <hip/hip_bf16.h>

typedef __attribute__((ext_vector_type(16))) _Float16 v16h;
typedef __attribute__((ext_vector_type(8)))  _Float16 v8h;
typedef __attribute__((ext_vector_type(4)))  _Float16 v4h;
typedef __attribute__((ext_vector_type(2)))  _Float16 v2h;
typedef __attribute__((ext_vector_type(8)))  float    v8f;

#define NWIN  4096
#define NTOK  49
#define CH    96
#define NHEAD 3
#define HDIM  32
#define QKV_D 288
#define SCALE 0.17677669529663689f   // 32^-0.5

// 16-bit WMMA A/B fragment for one lane: two contiguous 8-half (16B) chunks at
// p and p+16 halves (CDNA5 16-bit A/B layout; caller folds (lane>>4)*8 into p).
__device__ __forceinline__ v16h ldfrag(const _Float16* p) {
  union { v16h v; v8h h[2]; } u;
  u.h[0] = *(const v8h*)(p);
  u.h[1] = *(const v8h*)(p + 16);
  return u.v;
}

__device__ __forceinline__ v8f wmma_f16(v16h a, v16h b, v8f c) {
  return __builtin_amdgcn_wmma_f32_16x16x32_f16(false, a, false, b, (short)0, c,
                                                false, false);
}

extern "C" __global__ void convert_weights(const float* __restrict__ qkv_w,
                                           const float* __restrict__ proj_w,
                                           _Float16* __restrict__ ws) {
  int i = blockIdx.x * 256 + threadIdx.x;
  if (i < QKV_D * CH) ws[i] = (_Float16)qkv_w[i];
  if (i < CH * CH)    ws[QKV_D * CH + i] = (_Float16)proj_w[i];
}

extern "C" __global__ __launch_bounds__(128)
void swin_window_attn(const float* __restrict__ x,
                      const float* __restrict__ maskp,
                      const float* __restrict__ qkv_b,
                      const float* __restrict__ proj_b,
                      const _Float16* __restrict__ wq,   // [288][96] f16
                      const _Float16* __restrict__ wp,   // [96][96]  f16
                      float* __restrict__ out) {
  extern __shared__ _Float16 smem[];
  _Float16* xs  = smem;           // [64][96] x f16
  _Float16* qs  = smem + 6144;    // [3][64][32] q*scale
  _Float16* ks  = smem + 12288;   // [3][64][32] k
  _Float16* vts = smem + 18432;   // [3][32][64] v transposed
  _Float16* ps  = smem + 24576;   // [64][64] attn probs
  _Float16* os  = smem + 28672;   // [64][96] attn output

  const int blk  = blockIdx.x;
  const int wi   = blk & 63;              // mask window index (b % nW)
  const int tid  = threadIdx.x;
  const int lane = tid & 31;
  // Wave id as an SGPR: keeps nt-loops & which-branches uniform (scalar
  // branches, no exec masking around WMMA).
  const int w    = __builtin_amdgcn_readfirstlane(tid >> 5);
  const int lh   = lane & 15;             // tile col / row selector
  const int lg2  = lane >> 4;             // half-wave

  // ---- stage 0: x -> LDS f16 (vectorized), rows >=49 zero-padded ----
  {
    const float4* xb4 = (const float4*)(x + (size_t)blk * (NTOK * CH));
    for (int i = tid; i < 64 * 24; i += 128) {   // 24 float4 per 96-col row
      int m = i / 24, c4 = i % 24;
      float4 v = (m < NTOK) ? xb4[m * 24 + c4] : make_float4(0.f, 0.f, 0.f, 0.f);
      v4h h;
      h[0] = (_Float16)v.x; h[1] = (_Float16)v.y;
      h[2] = (_Float16)v.z; h[3] = (_Float16)v.w;
      *(v4h*)&xs[m * CH + c4 * 4] = h;
    }
  }
  __syncthreads();

  // ---- stage 1: QKV = x * Wqkv^T + b ----
  // N-split across waves: each weight B-fragment is loaded ONCE per block and
  // reused across all 4 M-tiles held in registers.
  {
    v16h ax[4][3];
#pragma unroll
    for (int mt = 0; mt < 4; ++mt)
#pragma unroll
      for (int kt = 0; kt < 3; ++kt)
        ax[mt][kt] = ldfrag(xs + (mt * 16 + lh) * CH + kt * 32 + lg2 * 8);

#pragma unroll
    for (int i = 0; i < 5; ++i) {
      const int nt = w + 4 * i;           // uniform (SGPR) tile index
      if (nt < 18) {
        v16h bw[3];
#pragma unroll
        for (int kt = 0; kt < 3; ++kt)
          bw[kt] = ldfrag(wq + (nt * 16 + lh) * CH + kt * 32 + lg2 * 8);
        float bias = qkv_b[nt * 16 + lh];
        const int which = nt / 6;         // 0=q 1=k 2=v (scalar)
        const int h     = (nt % 6) >> 1;
        const int hd    = ((nt & 1) << 4) + lh;
#pragma unroll
        for (int mt = 0; mt < 4; ++mt) {
          v8f c = {};
#pragma unroll
          for (int kt = 0; kt < 3; ++kt) c = wmma_f16(ax[mt][kt], bw[kt], c);
          const int mbase = mt * 16 + lg2 * 8;
          if (which == 0) {
#pragma unroll
            for (int r = 0; r < 8; ++r)
              qs[(h * 64 + mbase + r) * HDIM + hd] =
                  (_Float16)((c[r] + bias) * SCALE);
          } else if (which == 1) {
#pragma unroll
            for (int r = 0; r < 8; ++r)
              ks[(h * 64 + mbase + r) * HDIM + hd] = (_Float16)(c[r] + bias);
          } else {
            // transposed V: consecutive r contiguous -> packed b32 stores
#pragma unroll
            for (int r2 = 0; r2 < 4; ++r2) {
              v2h pr;
              pr[0] = (_Float16)(c[2 * r2] + bias);
              pr[1] = (_Float16)(c[2 * r2 + 1] + bias);
              *(v2h*)&vts[(h * HDIM + hd) * 64 + mbase + 2 * r2] = pr;
            }
          }
        }
      }
    }
  }
  __syncthreads();   // q/k/vT consumed across waves below

  // ---- stage 2: per-head attention (M-strip per wave, no global weights) ----
  const float* mb = maskp + (size_t)wi * (NTOK * NTOK);
  for (int h = 0; h < NHEAD; ++h) {
    v16h aq = ldfrag(qs + (h * 64 + w * 16 + lh) * HDIM + lg2 * 8);
    v8f cc[4];
#pragma unroll
    for (int nt = 0; nt < 4; ++nt) {
      v16h bk = ldfrag(ks + (h * 64 + nt * 16 + lh) * HDIM + lg2 * 8);
      v8f z = {};
      cc[nt] = wmma_f16(aq, bk, z);
    }
    // mask add + softmax over the 49 real columns
#pragma unroll
    for (int r = 0; r < 8; ++r) {
      int rr = w * 16 + lg2 * 8 + r;
      float lgv[4];
#pragma unroll
      for (int nt = 0; nt < 4; ++nt) {
        int col = nt * 16 + lh;
        lgv[nt] = (rr < NTOK && col < NTOK) ? (cc[nt][r] + mb[rr * NTOK + col])
                                            : -3.0e38f;
      }
      float mx = fmaxf(fmaxf(lgv[0], lgv[1]), fmaxf(lgv[2], lgv[3]));
#pragma unroll
      for (int s = 1; s <= 8; s <<= 1) mx = fmaxf(mx, __shfl_xor(mx, s));
      mx = fmaxf(mx, -1.0e30f);
      float sum = 0.0f;
#pragma unroll
      for (int nt = 0; nt < 4; ++nt) {
        float p = __expf(lgv[nt] - mx);
        lgv[nt] = p;
        sum += p;
      }
#pragma unroll
      for (int s = 1; s <= 8; s <<= 1) sum += __shfl_xor(sum, s);
      float inv = 1.0f / fmaxf(sum, 1.0e-30f);
#pragma unroll
      for (int nt = 0; nt < 4; ++nt)
        ps[rr * 64 + nt * 16 + lh] = (_Float16)(lgv[nt] * inv);
    }
    // out_h = P @ V  (own rows of ps -> no barrier needed inside head loop)
    v16h pa[2];
#pragma unroll
    for (int kt = 0; kt < 2; ++kt)
      pa[kt] = ldfrag(ps + (w * 16 + lh) * 64 + kt * 32 + lg2 * 8);
#pragma unroll
    for (int nt = 0; nt < 2; ++nt) {
      v8f c = {};
#pragma unroll
      for (int kt = 0; kt < 2; ++kt) {
        v16h bv = ldfrag(vts + (h * HDIM + nt * 16 + lh) * 64 + kt * 32 + lg2 * 8);
        c = wmma_f16(pa[kt], bv, c);
      }
#pragma unroll
      for (int r = 0; r < 8; ++r) {
        int m = w * 16 + lg2 * 8 + r;
        os[m * CH + h * HDIM + nt * 16 + lh] = (_Float16)c[r];
      }
    }
  }
  __syncthreads();   // os consumed across waves in proj

  // ---- stage 3: proj = os * Wp^T + b (N-split, weights loaded once) ----
  {
    v16h ao[4][3];
#pragma unroll
    for (int mt = 0; mt < 4; ++mt)
#pragma unroll
      for (int kt = 0; kt < 3; ++kt)
        ao[mt][kt] = ldfrag(os + (mt * 16 + lh) * CH + kt * 32 + lg2 * 8);

    float* ob = out + (size_t)blk * (NTOK * CH);
#pragma unroll
    for (int i = 0; i < 2; ++i) {
      const int nt = w + 4 * i;           // uniform
      if (nt < 6) {
        v16h bw[3];
#pragma unroll
        for (int kt = 0; kt < 3; ++kt)
          bw[kt] = ldfrag(wp + (nt * 16 + lh) * CH + kt * 32 + lg2 * 8);
        float bias = proj_b[nt * 16 + lh];
#pragma unroll
        for (int mt = 0; mt < 4; ++mt) {
          v8f c = {};
#pragma unroll
          for (int kt = 0; kt < 3; ++kt) c = wmma_f16(ao[mt][kt], bw[kt], c);
#pragma unroll
          for (int r = 0; r < 8; ++r) {
            int m = mt * 16 + lg2 * 8 + r;
            if (m < NTOK) ob[m * CH + nt * 16 + lh] = c[r] + bias;
          }
        }
      }
    }
  }
}

extern "C" void kernel_launch(void* const* d_in, const int* in_sizes, int n_in,
                              void* d_out, int out_size, void* d_ws, size_t ws_size,
                              hipStream_t stream) {
  const float* x      = (const float*)d_in[0];
  const float* maskp  = (const float*)d_in[1];
  const float* qkv_w  = (const float*)d_in[2];
  const float* qkv_b  = (const float*)d_in[3];
  const float* proj_w = (const float*)d_in[4];
  const float* proj_b = (const float*)d_in[5];
  float* out = (float*)d_out;
  _Float16* ws = (_Float16*)d_ws;   // [288*96] qkv_w f16, then [96*96] proj_w f16

  convert_weights<<<(QKV_D * CH + 255) / 256, 256, 0, stream>>>(qkv_w, proj_w, ws);

  size_t ldsBytes = 34816 * sizeof(_Float16);   // 69,632 B (< 320 KB/WGP)
  swin_window_attn<<<NWIN, 128, ldsBytes, stream>>>(
      x, maskp, qkv_b, proj_b, ws, ws + QKV_D * CH, out);
}
